// HeteroNewsCompanyGNN_48696339202467
// MI455X (gfx1250) — matrix operand order repaired and checked by reading, not compiled, not run on hardware
//
#include <hip/hip_runtime.h>
#include <hip/hip_bf16.h>
#include <float.h>

typedef __attribute__((ext_vector_type(16))) _Float16 v16h;
typedef __attribute__((ext_vector_type(8)))  float    v8f;

union F16x16 { uint4 q[2]; _Float16 h[16]; v16h v; };

// ---------------------------------------------------------------------------
// utility
// ---------------------------------------------------------------------------
__global__ void fill_u32_kernel(unsigned* __restrict__ p, unsigned v, long n) {
  long i = (long)blockIdx.x * blockDim.x + threadIdx.x;
  if (i < n) p[i] = v;
}

// monotone float<->uint mapping so unsigned atomicMax == float max
__device__ __forceinline__ unsigned fkey(float f) {
  unsigned u = __float_as_uint(f);
  return (u & 0x80000000u) ? ~u : (u | 0x80000000u);
}
__device__ __forceinline__ float fdec(unsigned k) {
  unsigned u = (k & 0x80000000u) ? (k & 0x7fffffffu) : ~k;
  return __uint_as_float(u);
}

// ---------------------------------------------------------------------------
// WMMA GEMM: C[M,64] = act(A[M,K] @ W[K,64] + bias (+ C if accum))
// block = 256 threads (8 waves); each wave computes a 16x64 tile via 4 WMMAs
// per K-step of 32.  A staging has a branch-free fast path (interior tiles)
// and all LDS traffic is b128:
//   - sA row-major, row stride 40 halves (80B, 16B aligned, bank-spread)
//   - sW permuted per 32-K tile: sW[(ks*64+n)*32 + kk], so a lane's B
//     fragment (k = g*16+i) is 32 contiguous bytes.
// ---------------------------------------------------------------------------
#define GEMM_KMAX 416   // covers K=385 (news projection); others are <=128
#define SA_STRIDE 40    // halves per A-tile row (32 data + 8 pad)

__global__ __launch_bounds__(256) void gemm64_wmma(
    const float* __restrict__ A, int M, int K,
    const float* __restrict__ W, const float* __restrict__ bias,
    float* __restrict__ C, int accum, int do_relu)
{
  __shared__ _Float16 sW[GEMM_KMAX * 64];      // 52 KB (permuted)
  __shared__ _Float16 sA[128 * SA_STRIDE];     // 10 KB
  __shared__ float    sB[64];

  const int tid  = threadIdx.x;
  const int wv   = tid >> 5;      // wave id 0..7
  const int lane = tid & 31;
  const int g    = lane >> 4;     // half-wave group
  const int lm   = lane & 15;
  const int rowBase = blockIdx.x * 128;
  const int kSteps  = (K + 31) >> 5;
  const int Kpad    = kSteps << 5;

  // stage W (f16, permuted) and bias once -- branch-free (clamped address)
  for (int i = tid; i < Kpad * 64; i += 256) {
    int k = i >> 6, n = i & 63;
    int ks = k >> 5, kk = k & 31;
    float v = W[(long)(k < K ? k : K - 1) * 64 + n];
    sW[((ks << 6) + n) * 32 + kk] = (_Float16)(k < K ? v : 0.0f);
  }
  if (tid < 64) sB[tid] = bias ? bias[tid] : 0.0f;

  v8f acc[4] = {};
  // A-tile staging geometry: each thread owns half a row (16 halves = 32B)
  const int sr  = tid >> 1;            // staged row 0..127
  const int kh  = (tid & 1) << 4;      // 0 or 16 within the 32-K tile
  const int gr  = rowBase + sr;
  const float* __restrict__ Arow = A + (long)(gr < M ? gr : M - 1) * K;
  const bool rowOK  = gr < M;
  const int arow = wv * 16 + lm;

  for (int ks = 0; ks < kSteps; ++ks) {
    const int kb = ks << 5;
    __syncthreads();  // protect sA WAR from previous iteration (and sW stage)
    // stage 128x32 A tile
    {
      F16x16 buf;
      const int kbase = kb + kh;
      if (rowOK && (kbase + 16 <= K)) {
        // fast path: 16 unconditional loads, results used unconditionally
#pragma unroll
        for (int c = 0; c < 16; ++c)
          buf.h[c] = (_Float16)Arow[kbase + c];
      } else {
        // tail/edge path: clamped address + value select
#pragma unroll
        for (int c = 0; c < 16; ++c) {
          int gk = kbase + c;
          float v = Arow[gk < K ? gk : K - 1];
          buf.h[c] = (_Float16)((rowOK && gk < K) ? v : 0.0f);
        }
      }
      *(uint4*)(sA + sr * SA_STRIDE + kh)     = buf.q[0];
      *(uint4*)(sA + sr * SA_STRIDE + kh + 8) = buf.q[1];
    }
    __syncthreads();

    // A fragment: halves k = g*16+(0..15) -> two b128 loads
    F16x16 af;
    af.q[0] = *(const uint4*)(sA + arow * SA_STRIDE + g * 16);
    af.q[1] = *(const uint4*)(sA + arow * SA_STRIDE + g * 16 + 8);

#pragma unroll
    for (int t = 0; t < 4; ++t) {
      // B fragment: k = g*16 + i (i=0..15) for column n = t*16+lm -> 32B run
      F16x16 bfr;
      const uint4* bp = (const uint4*)(sW + ((size_t)((ks << 6) + t * 16 + lm) * 32) + (g << 4));
      bfr.q[0] = bp[0];
      bfr.q[1] = bp[1];
      acc[t] = __builtin_amdgcn_wmma_f32_16x16x32_f16(
          false, af.v, false, bfr.v, (short)0, acc[t], false, false);
    }
  }

  // store: C/D layout -> VGPR r holds row (r + 8*g), lane lm = col
  const bool fullM = (rowBase + 128) <= M;
  if (fullM) {
#pragma unroll
    for (int t = 0; t < 4; ++t) {
#pragma unroll
      for (int r = 0; r < 8; ++r) {
        long idx = (long)(rowBase + wv * 16 + g * 8 + r) * 64 + t * 16 + lm;
        float val = acc[t][r] + sB[t * 16 + lm];
        if (accum) val += C[idx];
        if (do_relu) val = fmaxf(val, 0.0f);
        C[idx] = val;
      }
    }
  } else {
#pragma unroll
    for (int t = 0; t < 4; ++t) {
#pragma unroll
      for (int r = 0; r < 8; ++r) {
        int row = rowBase + wv * 16 + g * 8 + r;
        if (row < M) {
          long idx = (long)row * 64 + t * 16 + lm;
          float val = acc[t][r] + sB[t * 16 + lm];
          if (accum) val += C[idx];
          if (do_relu) val = fmaxf(val, 0.0f);
          C[idx] = val;
        }
      }
    }
  }
}

// ---------------------------------------------------------------------------
// attention edge kernels (softmax over incoming edges per company)
// ---------------------------------------------------------------------------
__global__ void att_score_kernel(const float* __restrict__ news_a,
                                 const float* __restrict__ comp_a,
                                 const float* __restrict__ va,
                                 const int* __restrict__ src,
                                 const int* __restrict__ dst,
                                 float* __restrict__ scores,
                                 unsigned* __restrict__ mkey, int E)
{
  __shared__ float4 sva[16];
  if (threadIdx.x < 16) sva[threadIdx.x] = ((const float4*)va)[threadIdx.x];
  __syncthreads();
  int e = blockIdx.x * blockDim.x + threadIdx.x;
  if (e >= E) return;
  const float4* na = (const float4*)(news_a + (long)src[e] * 64);
  const float4* ca = (const float4*)(comp_a + (long)dst[e] * 64);
  float s = 0.f;
#pragma unroll
  for (int q = 0; q < 16; ++q) {
    float4 a = na[q], c = ca[q], v = sva[q];
    s += tanhf(a.x + c.x) * v.x + tanhf(a.y + c.y) * v.y +
         tanhf(a.z + c.z) * v.z + tanhf(a.w + c.w) * v.w;
  }
  scores[e] = s;
  atomicMax(mkey + dst[e], fkey(s));
}

__global__ void att_norm_kernel(float* __restrict__ scores,
                                const unsigned* __restrict__ mkey,
                                float* __restrict__ denom,
                                const int* __restrict__ dst, int E)
{
  int e = blockIdx.x * blockDim.x + threadIdx.x;
  if (e >= E) return;
  int d = dst[e];
  float ex = expf(scores[e] - fdec(mkey[d]));
  scores[e] = ex;
  atomicAdd(denom + d, ex);
}

__global__ void att_apply_kernel(const float* __restrict__ news_h,
                                 const float* __restrict__ scores,
                                 const float* __restrict__ denom,
                                 const int* __restrict__ src,
                                 const int* __restrict__ dst,
                                 float* __restrict__ agg, long E64)
{
  long t = (long)blockIdx.x * blockDim.x + threadIdx.x;
  if (t >= E64) return;
  int e = (int)(t >> 6), j = (int)(t & 63);
  int d = dst[e];
  float w = scores[e] / fmaxf(denom[d], 1e-9f);
  atomicAdd(agg + (long)d * 64 + j, news_h[(long)src[e] * 64 + j] * w);
}

// ---------------------------------------------------------------------------
// SAGE mean aggregation (comp->comp edges)
// ---------------------------------------------------------------------------
__global__ void scatter_sum_kernel(const float* __restrict__ x,
                                   const int* __restrict__ src,
                                   const int* __restrict__ dst,
                                   float* __restrict__ sum,
                                   float* __restrict__ cnt, long E64)
{
  long t = (long)blockIdx.x * blockDim.x + threadIdx.x;
  if (t >= E64) return;
  int e = (int)(t >> 6), j = (int)(t & 63);
  int d = dst[e];
  if (j == 0) atomicAdd(cnt + d, 1.0f);
  atomicAdd(sum + (long)d * 64 + j, x[(long)src[e] * 64 + j]);
}

__global__ void mean_div_kernel(float* __restrict__ sum,
                                const float* __restrict__ cnt, long n64)
{
  long i = (long)blockIdx.x * blockDim.x + threadIdx.x;
  if (i < n64) sum[i] /= fmaxf(cnt[i >> 6], 1.0f);
}

// ---------------------------------------------------------------------------
// LayerNorm over H=64, one wave32 per row (2 elems/lane), shfl reductions
// ---------------------------------------------------------------------------
__global__ __launch_bounds__(256) void layernorm_kernel(
    float* __restrict__ x, const float* __restrict__ g,
    const float* __restrict__ b, int R)
{
  int wv = threadIdx.x >> 5, lane = threadIdx.x & 31;
  int row = blockIdx.x * 8 + wv;
  if (row >= R) return;
  long base = (long)row * 64;
  float x0 = x[base + lane], x1 = x[base + 32 + lane];
  float s = x0 + x1;
  for (int o = 16; o > 0; o >>= 1) s += __shfl_xor(s, o, 32);
  float mu = s * (1.0f / 64.0f);
  float d0 = x0 - mu, d1 = x1 - mu;
  float v = d0 * d0 + d1 * d1;
  for (int o = 16; o > 0; o >>= 1) v += __shfl_xor(v, o, 32);
  float r = rsqrtf(v * (1.0f / 64.0f) + 1e-5f);
  x[base + lane]      = d0 * r * g[lane]      + b[lane];
  x[base + 32 + lane] = d1 * r * g[lane + 32] + b[lane + 32];
}

// ---------------------------------------------------------------------------
// classifier head: logits = relu(x@Wc1+bc1)@Wc2+bc2, one thread per row
// ---------------------------------------------------------------------------
__global__ __launch_bounds__(256) void head_kernel(
    const float* __restrict__ x, const float* __restrict__ Wc1,
    const float* __restrict__ bc1, const float* __restrict__ Wc2,
    const float* __restrict__ bc2, float* __restrict__ out, int R)
{
  __shared__ float sW1[64 * 32];
  __shared__ float sb1[32];
  __shared__ float sW2[32];
  __shared__ float sb2;
  for (int i = threadIdx.x; i < 2048; i += 256) sW1[i] = Wc1[i];
  if (threadIdx.x < 32) { sb1[threadIdx.x] = bc1[threadIdx.x]; sW2[threadIdx.x] = Wc2[threadIdx.x]; }
  if (threadIdx.x == 0) sb2 = bc2[0];
  __syncthreads();
  int r = blockIdx.x * 256 + threadIdx.x;
  if (r >= R) return;
  float xr[64];
#pragma unroll
  for (int k = 0; k < 64; ++k) xr[k] = x[(long)r * 64 + k];
  float o = sb2;
#pragma unroll 4
  for (int jj = 0; jj < 32; ++jj) {
    float s = sb1[jj];
#pragma unroll
    for (int k = 0; k < 64; ++k) s += xr[k] * sW1[k * 32 + jj];
    o += fmaxf(s, 0.0f) * sW2[jj];
  }
  out[r] = o;
}

// ---------------------------------------------------------------------------
// orchestration
// ---------------------------------------------------------------------------
extern "C" void kernel_launch(void* const* d_in, const int* in_sizes, int n_in,
                              void* d_out, int out_size, void* d_ws, size_t ws_size,
                              hipStream_t stream)
{
  const float* news_x    = (const float*)d_in[0];
  const float* company_x = (const float*)d_in[1];
  const float* Wp_news   = (const float*)d_in[2];
  const float* bp_news   = (const float*)d_in[3];
  const float* Wp_comp   = (const float*)d_in[4];
  const float* bp_comp   = (const float*)d_in[5];
  const float* Wa_news   = (const float*)d_in[6];
  const float* Wa_comp   = (const float*)d_in[7];
  const float* va        = (const float*)d_in[8];
  const float* Wf        = (const float*)d_in[9];
  const float* bfv       = (const float*)d_in[10];
  const float* ln_f_g    = (const float*)d_in[11];
  const float* ln_f_b    = (const float*)d_in[12];
  // 13..15: W1n_* (dead news branch)  16..18: comp conv1  19..21: dead  22..24: comp conv2
  const float* W1c_l = (const float*)d_in[16];
  const float* b1c   = (const float*)d_in[17];
  const float* W1c_r = (const float*)d_in[18];
  const float* W2c_l = (const float*)d_in[22];
  const float* b2c   = (const float*)d_in[23];
  const float* W2c_r = (const float*)d_in[24];
  const float* ln_comp_g = (const float*)d_in[27];
  const float* ln_comp_b = (const float*)d_in[28];
  const float* Wc1 = (const float*)d_in[29];
  const float* bc1 = (const float*)d_in[30];
  const float* Wc2 = (const float*)d_in[31];
  const float* bc2 = (const float*)d_in[32];
  const int* m_src = (const int*)d_in[33];
  const int* m_dst = (const int*)d_in[34];
  const int* c_src = (const int*)d_in[37];
  const int* c_dst = (const int*)d_in[38];

  const int K_news = in_sizes[2] / 64;       // 385
  const int N_news = in_sizes[0] / K_news;   // 200000
  const int K_comp = in_sizes[4] / 64;       // 24
  const int N_comp = in_sizes[1] / K_comp;   // 5000
  const int E_M    = in_sizes[33];           // 1M mention edges
  const int E_CC   = in_sizes[37];           // 100K comp edges

  // workspace carve-out (~113 MB)
  char* ws = (char*)d_ws;
  size_t off = 0;
  auto carve = [&](size_t bytes) -> void* {
    void* p = ws + off; off += (bytes + 255) & ~(size_t)255; return p;
  };
  float*    news_h   = (float*)carve((size_t)N_news * 64 * 4);
  float*    news_a   = (float*)carve((size_t)N_news * 64 * 4);
  float*    comp_h   = (float*)carve((size_t)N_comp * 64 * 4);
  float*    comp_a   = (float*)carve((size_t)N_comp * 64 * 4);
  float*    comp_h2  = (float*)carve((size_t)N_comp * 64 * 4);
  float*    comp_1   = (float*)carve((size_t)N_comp * 64 * 4);
  float*    comp_2   = (float*)carve((size_t)N_comp * 64 * 4);
  float*    comp_sum = (float*)carve((size_t)N_comp * 64 * 4);
  float*    comp_cnt = (float*)carve((size_t)N_comp * 4);
  float*    news_agg = (float*)carve((size_t)N_comp * 64 * 4);
  float*    scores   = (float*)carve((size_t)E_M * 4);
  unsigned* mkey     = (unsigned*)carve((size_t)N_comp * 4);
  float*    denom    = (float*)carve((size_t)N_comp * 4);
  (void)ws_size; (void)n_in; (void)out_size;

  auto g1 = [](long n) { return dim3((unsigned)((n + 255) / 256)); };
  dim3 b256(256);

  // 1-2: projections (relu)
  gemm64_wmma<<<(N_news + 127) / 128, b256, 0, stream>>>(news_x, N_news, K_news, Wp_news, bp_news, news_h, 0, 1);
  gemm64_wmma<<<(N_comp + 127) / 128, b256, 0, stream>>>(company_x, N_comp, K_comp, Wp_comp, bp_comp, comp_h, 0, 1);
  // 3: attention pre-transforms
  gemm64_wmma<<<(N_news + 127) / 128, b256, 0, stream>>>(news_h, N_news, 64, Wa_news, nullptr, news_a, 0, 0);
  gemm64_wmma<<<(N_comp + 127) / 128, b256, 0, stream>>>(comp_h, N_comp, 64, Wa_comp, nullptr, comp_a, 0, 0);
  // 4: edge softmax attention -> news_agg
  fill_u32_kernel<<<g1(N_comp), b256, 0, stream>>>(mkey, 0x007FFFFFu, N_comp);           // fkey(-inf)
  fill_u32_kernel<<<g1(N_comp), b256, 0, stream>>>((unsigned*)denom, 0u, N_comp);
  fill_u32_kernel<<<g1((long)N_comp * 64), b256, 0, stream>>>((unsigned*)news_agg, 0u, (long)N_comp * 64);
  att_score_kernel<<<g1(E_M), b256, 0, stream>>>(news_a, comp_a, va, m_src, m_dst, scores, mkey, E_M);
  att_norm_kernel<<<g1(E_M), b256, 0, stream>>>(scores, mkey, denom, m_dst, E_M);
  att_apply_kernel<<<g1((long)E_M * 64), b256, 0, stream>>>(news_h, scores, denom, m_src, m_dst, news_agg, (long)E_M * 64);
  // 5: fused update: LN(relu([comp_h | news_agg] @ Wf + bf))
  gemm64_wmma<<<(N_comp + 127) / 128, b256, 0, stream>>>(comp_h, N_comp, 64, Wf, nullptr, comp_h2, 0, 0);
  gemm64_wmma<<<(N_comp + 127) / 128, b256, 0, stream>>>(news_agg, N_comp, 64, Wf + 64 * 64, bfv, comp_h2, 1, 1);
  layernorm_kernel<<<(N_comp + 7) / 8, b256, 0, stream>>>(comp_h2, ln_f_g, ln_f_b, N_comp);
  // 6: comp SAGE conv1 + LN
  fill_u32_kernel<<<g1((long)N_comp * 64), b256, 0, stream>>>((unsigned*)comp_sum, 0u, (long)N_comp * 64);
  fill_u32_kernel<<<g1(N_comp), b256, 0, stream>>>((unsigned*)comp_cnt, 0u, N_comp);
  scatter_sum_kernel<<<g1((long)E_CC * 64), b256, 0, stream>>>(comp_h2, c_src, c_dst, comp_sum, comp_cnt, (long)E_CC * 64);
  mean_div_kernel<<<g1((long)N_comp * 64), b256, 0, stream>>>(comp_sum, comp_cnt, (long)N_comp * 64);
  gemm64_wmma<<<(N_comp + 127) / 128, b256, 0, stream>>>(comp_sum, N_comp, 64, W1c_l, b1c, comp_1, 0, 0);
  gemm64_wmma<<<(N_comp + 127) / 128, b256, 0, stream>>>(comp_h2, N_comp, 64, W1c_r, nullptr, comp_1, 1, 1);
  layernorm_kernel<<<(N_comp + 7) / 8, b256, 0, stream>>>(comp_1, ln_comp_g, ln_comp_b, N_comp);
  // 7: comp SAGE conv2 + LN
  fill_u32_kernel<<<g1((long)N_comp * 64), b256, 0, stream>>>((unsigned*)comp_sum, 0u, (long)N_comp * 64);
  fill_u32_kernel<<<g1(N_comp), b256, 0, stream>>>((unsigned*)comp_cnt, 0u, N_comp);
  scatter_sum_kernel<<<g1((long)E_CC * 64), b256, 0, stream>>>(comp_1, c_src, c_dst, comp_sum, comp_cnt, (long)E_CC * 64);
  mean_div_kernel<<<g1((long)N_comp * 64), b256, 0, stream>>>(comp_sum, comp_cnt, (long)N_comp * 64);
  gemm64_wmma<<<(N_comp + 127) / 128, b256, 0, stream>>>(comp_sum, N_comp, 64, W2c_l, b2c, comp_2, 0, 0);
  gemm64_wmma<<<(N_comp + 127) / 128, b256, 0, stream>>>(comp_1, N_comp, 64, W2c_r, nullptr, comp_2, 1, 1);
  layernorm_kernel<<<(N_comp + 7) / 8, b256, 0, stream>>>(comp_2, ln_comp_g, ln_comp_b, N_comp);
  // 8: head
  head_kernel<<<g1(N_comp), b256, 0, stream>>>(comp_2, Wc1, bc1, Wc2, bc2, (float*)d_out, N_comp);
}
